// RelativeMultiheadAttention_25134148616677
// MI455X (gfx1250) — matrix operand order repaired and checked by reading, not compiled, not run on hardware
//
#include <hip/hip_runtime.h>
#include <hip/hip_bf16.h>
#include <stdint.h>

typedef __attribute__((ext_vector_type(16))) __bf16 v16bf;
typedef __attribute__((ext_vector_type(8)))  float  v8f;
typedef __attribute__((ext_vector_type(4)))  unsigned int v4u;
typedef __attribute__((ext_vector_type(8)))  int v8i;
typedef __attribute__((ext_vector_type(4)))  int v4i;

#define D_MODEL 1024
#define NHEAD   16
#define DK      64
#define SEQ     2048
#define BATCH   4
#define NROWS   (BATCH * SEQ)   // 8192
#define MAXREL  1024

__device__ __forceinline__ unsigned short f32_to_bf16_bits(float f) {
  union { float f; uint32_t u; } c; c.f = f;
  uint32_t u = c.u;
  return (unsigned short)((u + 0x7FFFu + ((u >> 16) & 1u)) >> 16);
}

union FragB16 { uint4 q[2]; v16bf v; unsigned short s[16]; };

// ---------------------------------------------------------------- TDM helper
// Issue a 2D tensor_load_to_lds: tile_d1 rows x tile_d0 elements (bf16, 2B),
// global row stride = stride_elems. Optional LDS padding (pad codes per ISA:
// interval 3 -> every 16 DWORDs, amount 3 -> 4 DWORDs).
__device__ __forceinline__ void tdm_load_2d(uint32_t lds_off, const void* gptr,
                                            uint32_t tile_d0, uint32_t tile_d1,
                                            uint32_t tensor_d0, uint32_t tensor_d1,
                                            uint32_t stride_elems, int pad) {
  uint64_t ga = (uint64_t)(uintptr_t)gptr;
  v4u g0;
  g0[0] = 1u;                                   // count=1, user descriptor
  g0[1] = lds_off;                              // LDS byte address
  g0[2] = (uint32_t)ga;                         // global_addr[31:0]
  g0[3] = (uint32_t)((ga >> 32) & 0x01FFFFFFu)  // global_addr[56:32]
          | 0x80000000u;                        // type=2 ("image") in [127:126]
  v8i g1;
  uint32_t w0 = (1u << 16);                     // data_size=1 -> 2 bytes
  if (pad) w0 |= (1u << 20) | (3u << 22) | (3u << 25); // pad_en, every 16 DW, +4 DW
  g1[0] = (int)w0;
  g1[1] = (int)((tensor_d0 & 0xFFFFu) << 16);                       // dim0[15:0]
  g1[2] = (int)(((tensor_d0 >> 16) & 0xFFFFu) |
                ((tensor_d1 & 0xFFFFu) << 16));                     // dim0[31:16], dim1[15:0]
  g1[3] = (int)(((tensor_d1 >> 16) & 0xFFFFu) |
                ((tile_d0 & 0xFFFFu) << 16));                       // dim1[31:16], tile0
  g1[4] = (int)(tile_d1 & 0xFFFFu);                                 // tile1 (tile2=0)
  g1[5] = (int)stride_elems;                                        // dim0_stride[31:0]
  g1[6] = 0;
  g1[7] = 0;
  v4i gz4; gz4[0] = 0; gz4[1] = 0; gz4[2] = 0; gz4[3] = 0;
  v8i gz8;
  #pragma unroll
  for (int i = 0; i < 8; ++i) gz8[i] = 0;
  // 6-arg form (clang-23 / therock-10.0 headers)
  __builtin_amdgcn_tensor_load_to_lds(g0, g1, gz4, gz4, gz8, 0);
}

// ---------------------------------------------------------------- convert
__global__ void k_f32_to_bf16(const float* __restrict__ src,
                              unsigned short* __restrict__ dst, int n) {
  int i = blockIdx.x * blockDim.x + threadIdx.x;
  if (i < n) dst[i] = f32_to_bf16_bits(src[i]);
}

// ---------------------------------------------------------------- GEMM (bf16 WMMA)
// C[M,N] = A[M,K] @ B[K,N] + bias.  mode 0: write fp32 Cout.
// mode 1: scatter columns into Q (scaled 1/8), K, V bf16 buffers.
__global__ __launch_bounds__(256)
void k_gemm_bf16(const unsigned short* __restrict__ A,
                 const unsigned short* __restrict__ Bm,
                 const float* __restrict__ bias,
                 int M, int N, int K, int mode,
                 float* __restrict__ Cout,
                 unsigned short* __restrict__ Qo,
                 unsigned short* __restrict__ Ko,
                 unsigned short* __restrict__ Vo)
{
  __shared__ alignas(16) unsigned short As[64][40];    // TDM-padded: 32 data + 8 pad
  __shared__ alignas(16) unsigned short Bs[128][40];   // transposed: [n][k]

  const int tid  = threadIdx.x;
  const int wave = tid >> 5;
  const int lane = tid & 31;
  const int l15  = lane & 15, lh = lane >> 4;
  const int m0 = blockIdx.y * 64;
  const int n0 = blockIdx.x * 128;
  const int wm = (wave >> 2) * 32;
  const int wn = (wave & 3) * 32;

  v8f acc[2][2];
  #pragma unroll
  for (int i = 0; i < 2; ++i)
    #pragma unroll
    for (int j = 0; j < 2; ++j)
      #pragma unroll
      for (int e = 0; e < 8; ++e) acc[i][j][e] = 0.f;

  const int bk = tid >> 3,  bnc = (tid & 7) * 16;   // B: 2x b128 per thread

  const uint32_t as_lds = (uint32_t)(uintptr_t)&As[0][0];

  for (int k0 = 0; k0 < K; k0 += 32) {
    // A tile 64x32 via Tensor Data Mover, padded to stride 40 in LDS
    if (wave == 0) {
      tdm_load_2d(as_lds, A + (size_t)m0 * K + k0,
                  /*tile*/32, 64, /*tensor*/32, 64, /*stride*/(uint32_t)K, /*pad*/1);
    }
    // B tile 32x128, transposed into LDS by all threads
    FragB16 bt;
    const unsigned short* bsrc = Bm + (size_t)(k0 + bk) * N + n0 + bnc;
    bt.q[0] = *(const uint4*)(bsrc);
    bt.q[1] = *(const uint4*)(bsrc + 8);
    if (k0 + 32 < K) __builtin_prefetch(bsrc + 32 * N, 0, 1);  // next k-panel
    #pragma unroll
    for (int i = 0; i < 16; ++i) Bs[bnc + i][bk] = bt.s[i];
    if (wave == 0) __builtin_amdgcn_s_wait_tensorcnt(0);
    __syncthreads();

    FragB16 af[2], bf[2];
    #pragma unroll
    for (int s = 0; s < 2; ++s) {   // A frags: row = lane&15, K chunks lh*8 / lh*8+16
      const unsigned short* p = &As[wm + s * 16 + l15][lh * 8];
      af[s].q[0] = *(const uint4*)(p);
      af[s].q[1] = *(const uint4*)(p + 16);
    }
    #pragma unroll
    for (int s = 0; s < 2; ++s) {   // B frags: col = lane&15, K chunks lh*8 / lh*8+16
      const unsigned short* p = &Bs[wn + s * 16 + l15][lh * 8];
      bf[s].q[0] = *(const uint4*)(p);
      bf[s].q[1] = *(const uint4*)(p + 16);
    }
    #pragma unroll
    for (int i = 0; i < 2; ++i)
      #pragma unroll
      for (int j = 0; j < 2; ++j)
        acc[i][j] = __builtin_amdgcn_wmma_f32_16x16x32_bf16(
            false, af[i].v, false, bf[j].v, (short)0, acc[i][j], false, false);
    __syncthreads();
  }

  #pragma unroll
  for (int i = 0; i < 2; ++i)
    #pragma unroll
    for (int j = 0; j < 2; ++j)
      #pragma unroll
      for (int v = 0; v < 8; ++v) {
        int row = m0 + wm + i * 16 + lh * 8 + v;   // C layout: row = 8*(lane>>4)+v
        int col = n0 + wn + j * 16 + l15;
        float val = acc[i][j][v] + bias[col];
        if (mode == 0) {
          Cout[(size_t)row * N + col] = val;
        } else {
          int sel = col >> 10, cw = col & 1023;
          if (sel == 0)      Qo[(size_t)row * 1024 + cw] = f32_to_bf16_bits(val * 0.125f);
          else if (sel == 1) Ko[(size_t)row * 1024 + cw] = f32_to_bf16_bits(val);
          else               Vo[(size_t)row * 1024 + cw] = f32_to_bf16_bits(val);
        }
      }
}

// ---------------------------------------------------------------- fused flash attention
// One workgroup (4 waves) per (b, h, 64-query block). Each wave owns 16 rows.
__global__ __launch_bounds__(128)
void k_attn(const unsigned short* __restrict__ Qb,
            const unsigned short* __restrict__ Kb,
            const unsigned short* __restrict__ Vb,
            const float* __restrict__ relb,
            unsigned short* __restrict__ Ob)
{
  __shared__ alignas(16) unsigned short Ks[32][64];     // 32 keys x 64 d (TDM)
  __shared__ alignas(16) unsigned short VsT[64][40];    // transposed: [d][j]
  __shared__ alignas(16) unsigned short Ps[4][16][40];  // per-wave P scratch

  const int tid  = threadIdx.x;
  const int wave = tid >> 5, lane = tid & 31;
  const int l15  = lane & 15, lh = lane >> 4;
  const int bidx = blockIdx.y >> 4, h = blockIdx.y & 15;
  const int mblock = blockIdx.x * 64;
  const size_t rowoff = (size_t)bidx * SEQ;
  const int qr = mblock + wave * 16;

  // Q tile 16x64 as two A-fragments (Q already scaled by 1/sqrt(dk))
  FragB16 qf[2];
  {
    const unsigned short* qpb = Qb + (rowoff + qr + l15) * D_MODEL + h * DK;
    #pragma unroll
    for (int f = 0; f < 2; ++f) {
      qf[f].q[0] = *(const uint4*)(qpb + f * 32 + lh * 8);
      qf[f].q[1] = *(const uint4*)(qpb + f * 32 + lh * 8 + 16);
    }
  }

  float mrun[8], lrun[8];
  #pragma unroll
  for (int v = 0; v < 8; ++v) { mrun[v] = -1e30f; lrun[v] = 0.f; }
  v8f oacc[4];
  #pragma unroll
  for (int n = 0; n < 4; ++n)
    #pragma unroll
    for (int e = 0; e < 8; ++e) oacc[n][e] = 0.f;

  const int cr = tid >> 2;          // 0..31  (key row)
  const int cc = (tid & 3) * 16;    // d chunk
  const uint32_t ks_lds = (uint32_t)(uintptr_t)&Ks[0][0];

  for (int j0 = 0; j0 < SEQ; j0 += 32) {
    // K tile 32x64 via Tensor Data Mover (row-major, contiguous d)
    if (wave == 0) {
      tdm_load_2d(ks_lds, Kb + (rowoff + j0) * D_MODEL + h * DK,
                  /*tile*/64, 32, /*tensor*/64, 32, /*stride*/D_MODEL, /*pad*/0);
    }
    // V tile staged transposed into LDS by all threads
    {
      const unsigned short* vs = Vb + (rowoff + j0 + cr) * D_MODEL + h * DK + cc;
      FragB16 vt;
      vt.q[0] = *(const uint4*)(vs);
      vt.q[1] = *(const uint4*)(vs + 8);
      #pragma unroll
      for (int i = 0; i < 16; ++i) VsT[cc + i][cr] = vt.s[i];
    }
    if (wave == 0) __builtin_amdgcn_s_wait_tensorcnt(0);
    __syncthreads();

    // S = Q @ K^T : two 16x16 tiles, K-dim 64 = two WMMA k-steps each
    v8f sac[2];
    #pragma unroll
    for (int nb = 0; nb < 2; ++nb) {
      #pragma unroll
      for (int e = 0; e < 8; ++e) sac[nb][e] = 0.f;
      #pragma unroll
      for (int f = 0; f < 2; ++f) {
        FragB16 kf;
        const unsigned short* p = &Ks[nb * 16 + l15][f * 32 + lh * 8];
        kf.q[0] = *(const uint4*)(p);
        kf.q[1] = *(const uint4*)(p + 16);
        sac[nb] = __builtin_amdgcn_wmma_f32_16x16x32_bf16(
            false, qf[f].v, false, kf.v, (short)0, sac[nb], false, false);
      }
    }

    // rel-bias + online softmax (rows split: lanes 0-15 <-> rows v, 16-31 <-> v+8)
    const int ibase = mblock + wave * 16 + lh * 8;
    #pragma unroll
    for (int v = 0; v < 8; ++v) {
      int ig = ibase + v;
      int r0 = ig - (j0 + l15) + MAXREL;
      int r1 = r0 - 16;
      r0 = r0 < 0 ? 0 : (r0 > 2 * MAXREL ? 2 * MAXREL : r0);
      r1 = r1 < 0 ? 0 : (r1 > 2 * MAXREL ? 2 * MAXREL : r1);
      float s0 = sac[0][v] + relb[r0 * NHEAD + h];
      float s1 = sac[1][v] + relb[r1 * NHEAD + h];
      float tmax = fmaxf(s0, s1);
      #pragma unroll
      for (int m = 1; m < 16; m <<= 1) tmax = fmaxf(tmax, __shfl_xor(tmax, m, 32));
      float mnew  = fmaxf(mrun[v], tmax);
      float scale = __expf(mrun[v] - mnew);
      float e0 = __expf(s0 - mnew), e1 = __expf(s1 - mnew);
      float rs = e0 + e1;
      #pragma unroll
      for (int m = 1; m < 16; m <<= 1) rs += __shfl_xor(rs, m, 32);
      lrun[v] = lrun[v] * scale + rs;
      mrun[v] = mnew;
      #pragma unroll
      for (int n = 0; n < 4; ++n) oacc[n][v] *= scale;
      Ps[wave][lh * 8 + v][l15]      = f32_to_bf16_bits(e0);  // C-layout -> LDS
      Ps[wave][lh * 8 + v][16 + l15] = f32_to_bf16_bits(e1);
    }

    // re-read P as an A-fragment (same-wave LDS ops are in-order)
    FragB16 pf;
    {
      const unsigned short* p = &Ps[wave][l15][lh * 8];
      pf.q[0] = *(const uint4*)(p);
      pf.q[1] = *(const uint4*)(p + 16);
    }
    // O += P @ V  (V fragments from transposed LDS: contiguous along j)
    #pragma unroll
    for (int nb = 0; nb < 4; ++nb) {
      FragB16 vf;
      const unsigned short* p = &VsT[nb * 16 + l15][lh * 8];
      vf.q[0] = *(const uint4*)(p);
      vf.q[1] = *(const uint4*)(p + 16);
      oacc[nb] = __builtin_amdgcn_wmma_f32_16x16x32_bf16(
          false, pf.v, false, vf.v, (short)0, oacc[nb], false, false);
    }
    __syncthreads();
  }

  // normalize and store bf16 into [B*S, D_MODEL] attn buffer
  #pragma unroll
  for (int v = 0; v < 8; ++v) {
    float inv = 1.f / lrun[v];
    int row = qr + lh * 8 + v;
    unsigned short* op = Ob + (rowoff + row) * D_MODEL + h * DK + l15;
    #pragma unroll
    for (int nb = 0; nb < 4; ++nb)
      op[nb * 16] = f32_to_bf16_bits(oacc[nb][v] * inv);
  }
}

// ---------------------------------------------------------------- launch
extern "C" void kernel_launch(void* const* d_in, const int* in_sizes, int n_in,
                              void* d_out, int out_size, void* d_ws, size_t ws_size,
                              hipStream_t stream) {
  const float* x  = (const float*)d_in[0];
  const float* Wq = (const float*)d_in[1];
  const float* bq = (const float*)d_in[2];
  const float* Wo = (const float*)d_in[3];
  const float* bo = (const float*)d_in[4];
  const float* rb = (const float*)d_in[5];
  (void)in_sizes; (void)n_in; (void)out_size; (void)ws_size;

  char* ws = (char*)d_ws;
  unsigned short* xb  = (unsigned short*)(ws);                       // 16 MB
  unsigned short* wqb = (unsigned short*)(ws + (size_t)16777216);    //  6 MB
  unsigned short* wob = (unsigned short*)(ws + (size_t)23068672);    //  2 MB
  unsigned short* Qb  = (unsigned short*)(ws + (size_t)25165824);    // 16 MB
  unsigned short* Kb  = (unsigned short*)(ws + (size_t)41943040);    // 16 MB
  unsigned short* Vb  = (unsigned short*)(ws + (size_t)58720256);    // 16 MB
  unsigned short* Ab  = (unsigned short*)(ws + (size_t)75497472);    // 16 MB

  const int nx  = NROWS * D_MODEL;
  const int nwq = D_MODEL * 3 * D_MODEL;
  const int nwo = D_MODEL * D_MODEL;
  k_f32_to_bf16<<<(nx  + 255) / 256, 256, 0, stream>>>(x,  xb,  nx);
  k_f32_to_bf16<<<(nwq + 255) / 256, 256, 0, stream>>>(Wq, wqb, nwq);
  k_f32_to_bf16<<<(nwo + 255) / 256, 256, 0, stream>>>(Wo, wob, nwo);

  dim3 g1(3 * D_MODEL / 128, NROWS / 64);   // QKV projection + scatter
  k_gemm_bf16<<<g1, 256, 0, stream>>>(xb, wqb, bq, NROWS, 3 * D_MODEL, D_MODEL, 1,
                                      nullptr, Qb, Kb, Vb);

  dim3 g2(SEQ / 64, BATCH * NHEAD);         // fused flash attention
  k_attn<<<g2, 128, 0, stream>>>(Qb, Kb, Vb, rb, Ab);

  dim3 g3(D_MODEL / 128, NROWS / 64);       // output projection
  k_gemm_bf16<<<g3, 256, 0, stream>>>(Ab, wob, bo, NROWS, D_MODEL, D_MODEL, 0,
                                      (float*)d_out, nullptr, nullptr, nullptr);
}
